// plainLSTM_72945724555713
// MI455X (gfx1250) — matrix operand order repaired and checked by reading, not compiled, not run on hardware
//
#include <hip/hip_runtime.h>
#include <hip/hip_bf16.h>

// ---------------------------------------------------------------------------
// LSTM  B=16, T=2048, D=1024, H=1024  for MI455X (gfx1250, wave32, WMMA)
//
// Phase 0: convert W_hh -> bf16 (L2-resident, 8MB), init h ping-pong, sync cnt
// Phase 1: xg = x @ W_ih^T + b_ih + b_hh   (bf16 WMMA GEMM, fp32 accum/out)
// Phase 2: persistent 16-workgroup scan; per step: broadcast h into LDS via
//          cluster async-load-to-LDS (ASYNCcnt), h@W_hh^T via WMMA,
//          block-local cell update, cluster-barrier (+atomic fallback) sync.
// ---------------------------------------------------------------------------

#define Bsz  16
#define Tlen 2048
#define Din  1024
#define Hdim 1024
#define G4   4096          // 4*H
#define M1   (Bsz * Tlen)  // 32768

typedef __bf16 v16bf __attribute__((ext_vector_type(16)));
typedef float  v8f   __attribute__((ext_vector_type(8)));

// K index inside a 16-bit A fragment (16x32), per CDNA5 ISA 7.12.2:
// lanes 0-15: VGPR0..3 -> K pairs {0,1},{2,3},{4,5},{6,7}; VGPR4..7 -> {16..23}
// lanes16-31: same + 8
static __device__ __forceinline__ int a_swz_k(int e, int half) {
    int j = e >> 1;
    int base = (j < 4) ? (2 * j) : (16 + 2 * (j - 4));
    return base + half * 8 + (e & 1);
}

// ---------------------------------------------------------------------------
// Phase 0: init
// ---------------------------------------------------------------------------
__global__ __launch_bounds__(256)
void lstm_init_kernel(const float* __restrict__ Whh, const float* __restrict__ h0,
                      __bf16* __restrict__ Whh_bf, __bf16* __restrict__ hg,
                      unsigned* __restrict__ cnt) {
    size_t i = (size_t)blockIdx.x * blockDim.x + threadIdx.x;
    if (i < (size_t)G4 * Hdim) Whh_bf[i] = (__bf16)Whh[i];
    if (i < (size_t)Bsz * Hdim) {
        __bf16 v = (__bf16)h0[i & (Hdim - 1)];   // broadcast h0 row over batch
        hg[i] = v;                                // buffer 0 (used at t=0)
        hg[(size_t)Bsz * Hdim + i] = v;           // buffer 1
    }
    if (i == 0) *cnt = 0u;
}

// ---------------------------------------------------------------------------
// Phase 1: xg[M1,4096] = x[M1,1024] @ W_ih^T + (b_ih + b_hh)
// 64x64 tile per block, 8 waves, 2 WMMA tiles per wave, K-step 32, bf16 LDS
// ---------------------------------------------------------------------------
#define TM  64
#define TN  64
#define TK  32
#define LDA (TK + 8)   // 40 elems -> 80B row stride: conflict-free 16-lane gathers

__global__ __launch_bounds__(256)
void gemm_xg_kernel(const float* __restrict__ x, const float* __restrict__ Wih,
                    const float* __restrict__ bih, const float* __restrict__ bhh,
                    float* __restrict__ xg) {
    __shared__ __bf16 As[TM * LDA];
    __shared__ __bf16 Bs[TN * LDA];

    const int tid  = threadIdx.x;
    const int lane = tid & 31;
    const int wv   = tid >> 5;
    const int half = lane >> 4;
    const int l15  = lane & 15;
    const int bm   = blockIdx.y;        // 0..511  (M blocks)
    const int bn   = blockIdx.x;        // 0..63   (N blocks)
    const int tm   = wv >> 1;           // 0..3 : wave's 16-row strip
    const int tn0  = (wv & 1) * 2;      // wave's two 16-col tiles: tn0, tn0+1

    v8f acc0 = {0.f,0.f,0.f,0.f,0.f,0.f,0.f,0.f};
    v8f acc1 = {0.f,0.f,0.f,0.f,0.f,0.f,0.f,0.f};

    const int r  = tid >> 2;            // 0..63  staging row
    const int cs = (tid & 3) * 8;       // 0,8,16,24 staging col segment

    for (int k0 = 0; k0 < Din; k0 += TK) {
        const float* xa = x   + (size_t)(bm * TM + r) * Din + k0 + cs;
        const float* wb = Wih + (size_t)(bn * TN + r) * Din + k0 + cs;
#pragma unroll
        for (int j = 0; j < 8; ++j) {
            As[r * LDA + cs + j] = (__bf16)xa[j];
            Bs[r * LDA + cs + j] = (__bf16)wb[j];
        }
        __syncthreads();

        v16bf a, b0, b1;
#pragma unroll
        for (int e = 0; e < 16; ++e) {
            a[e]  = As[(tm * 16 + l15) * LDA + a_swz_k(e, half)];
            int kb = a_swz_k(e, half);              // B frag: n=l15, k same as A
            b0[e] = Bs[(tn0 * 16 + l15) * LDA + kb];
            b1[e] = Bs[((tn0 + 1) * 16 + l15) * LDA + kb];
        }
        acc0 = __builtin_amdgcn_wmma_f32_16x16x32_bf16(false, a, false, b0,
                                                       (short)0, acc0, false, false);
        acc1 = __builtin_amdgcn_wmma_f32_16x16x32_bf16(false, a, false, b1,
                                                       (short)0, acc1, false, false);
        __syncthreads();
    }

    // epilogue: C layout -> VGPR rr holds M = rr + 8*half, N = l15
    const int row0  = bm * TM + tm * 16 + 8 * half;
    const int nc0   = bn * TN + tn0 * 16 + l15;
    const int nc1   = nc0 + 16;
    const float bi0 = bih[nc0] + bhh[nc0];
    const float bi1 = bih[nc1] + bhh[nc1];
#pragma unroll
    for (int rr = 0; rr < 8; ++rr) {
        size_t ro = (size_t)(row0 + rr) * G4;
        xg[ro + nc0] = acc0[rr] + bi0;
        xg[ro + nc1] = acc1[rr] + bi1;
    }
}

// ---------------------------------------------------------------------------
// Phase 2: persistent scan. 16 blocks x 256 threads. Block w owns h columns
// [w*64, w*64+64) and gate columns {g*1024 + w*64 .. +64} for g=0..3.
// ---------------------------------------------------------------------------
#define LDH 1032   // 2064B row stride -> 516 dwords, conflict-free A gathers

__global__ __launch_bounds__(256)
void lstm_scan_kernel(const float* __restrict__ xg, const __bf16* __restrict__ Whh,
                      const float* __restrict__ c0,
                      float* __restrict__ out, float* __restrict__ hN,
                      float* __restrict__ cN,
                      __bf16* __restrict__ hg, volatile unsigned* __restrict__ cnt) {
    __shared__ __bf16 hbuf[16 * LDH];        // staged h (16x1024 bf16)
    __shared__ float  gbuf[4][16][64];       // i,f,g,o pre-activations
    __shared__ float  cbuf[16 * 64];         // persistent cell state slice

    const int w    = blockIdx.x;             // 0..15
    const int tid  = threadIdx.x;
    const int lane = tid & 31;
    const int wv   = tid >> 5;
    const int half = lane >> 4;
    const int l15  = lane & 15;

    // wave handles tiles ti0, ti0+1 ; ti = gate*4 + sub
    const int ti0 = wv * 2, ti1 = ti0 + 1;
    const int g0 = ti0 >> 2, s0 = ti0 & 3;
    const int g1 = ti1 >> 2, s1 = ti1 & 3;
    const int gc0 = g0 * Hdim + w * 64 + s0 * 16 + l15;   // gate column / W_hh row
    const int gc1 = g1 * Hdim + w * 64 + s1 * 16 + l15;
    const __bf16* wrow0 = Whh + (size_t)gc0 * Hdim + half * 16;
    const __bf16* wrow1 = Whh + (size_t)gc1 * Hdim + half * 16;

    // multicast mask: broadcast to WG0..15 of the cluster, early-timeout bit 16.
    // (NOP-equivalent when not in a cluster: loads downgrade to plain global.)
    asm volatile("s_mov_b32 m0, %0" :: "s"(0x0001FFFFu));

    for (int idx = tid; idx < 16 * 64; idx += 256)
        cbuf[idx] = c0[w * 64 + (idx & 63)];
    __syncthreads();

    for (int t = 0; t < Tlen; ++t) {
        const __bf16* hgr = hg + (size_t)(t & 1) * (Bsz * Hdim);          // read h_t
        __bf16*       hgw = hg + (size_t)((t + 1) & 1) * (Bsz * Hdim);    // write h_{t+1}

        // Broadcast-stage full h (16x1024 bf16 = 32KB) into LDS via cluster
        // async load-to-LDS: no VGPR round trip, one L2 fetch multicast to all
        // 16 WGPs when cluster-launched. Per-lane LDS dest keeps padded layout.
        for (int idx = tid; idx < 2048; idx += 256) {
            int m = idx >> 7;
            int cseg = (idx & 127) * 8;                       // 8 bf16 = 16B
            unsigned ldsa = (unsigned)(size_t)(&hbuf[m * LDH + cseg]);
            const __bf16* gsrc = hgr + m * Hdim + cseg;
            asm volatile("cluster_load_async_to_lds_b128 %0, %1, off"
                         :: "v"(ldsa), "v"(gsrc) : "memory");
        }
        asm volatile("s_wait_asynccnt 0" ::: "memory");
        __syncthreads();

        v8f acc0 = {0.f,0.f,0.f,0.f,0.f,0.f,0.f,0.f};
        v8f acc1 = {0.f,0.f,0.f,0.f,0.f,0.f,0.f,0.f};
#pragma unroll 4
        for (int kt = 0; kt < Hdim / 32; ++kt) {
            v16bf a;
#pragma unroll
            for (int e = 0; e < 16; ++e)
                a[e] = hbuf[l15 * LDH + kt * 32 + half * 16 + e];
            v16bf b0 = *(const v16bf*)(wrow0 + kt * 32);   // L2-resident W_hh
            v16bf b1 = *(const v16bf*)(wrow1 + kt * 32);
            acc0 = __builtin_amdgcn_wmma_f32_16x16x32_bf16(false, a, false, b0,
                                                           (short)0, acc0, false, false);
            acc1 = __builtin_amdgcn_wmma_f32_16x16x32_bf16(false, a, false, b1,
                                                           (short)0, acc1, false, false);
        }

        // gates = WMMA acc + xg[t]; scatter into gbuf
#pragma unroll
        for (int rr = 0; rr < 8; ++rr) {
            int m = rr + 8 * half;                          // batch index
            size_t xrow = ((size_t)m * Tlen + t) * G4;
            gbuf[g0][m][s0 * 16 + l15] = acc0[rr] + xg[xrow + gc0];
            gbuf[g1][m][s1 * 16 + l15] = acc1[rr] + xg[xrow + gc1];
        }
        __syncthreads();

        // elementwise LSTM cell for this block's 16x64 slice
        for (int idx = tid; idx < 1024; idx += 256) {
            int m = idx >> 6, nl = idx & 63;
            float ig = 1.f / (1.f + __expf(-gbuf[0][m][nl]));
            float fg = 1.f / (1.f + __expf(-gbuf[1][m][nl]));
            float gg = tanhf(gbuf[2][m][nl]);
            float og = 1.f / (1.f + __expf(-gbuf[3][m][nl]));
            float c  = fg * cbuf[idx] + ig * gg;
            cbuf[idx] = c;
            float h  = og * tanhf(c);
            out[((size_t)m * Tlen + t) * Hdim + w * 64 + nl] = h;
            hgw[m * Hdim + w * 64 + nl] = (__bf16)h;
            if (t == Tlen - 1) {
                hN[m * Hdim + w * 64 + nl] = h;
                cN[m * Hdim + w * 64 + nl] = c;
            }
        }

        // prefetch next step's xg rows (global_prefetch_b8)
        if (t + 1 < Tlen && tid < Bsz)
            __builtin_prefetch(&xg[(((size_t)tid) * Tlen + t + 1) * G4 + w * 64], 0, 1);

        // ---- cross-workgroup step barrier ----
        __threadfence();                 // release: h_{t+1} stores visible
        __syncthreads();
        if (tid == 0) atomicAdd((unsigned*)cnt, 1u);
        if (wv == 0) __builtin_amdgcn_s_cluster_barrier();      // HW fast path
        else asm volatile("s_barrier_wait -3" ::: "memory");
        if (tid == 0) {                  // correctness fallback (non-cluster launch)
            unsigned tgt = 16u * (unsigned)(t + 1);
            while (*cnt < tgt) __builtin_amdgcn_s_sleep(1);
        }
        __syncthreads();
        __threadfence();                 // acquire: drop stale WGP$ lines
    }
}

// ---------------------------------------------------------------------------
extern "C" void kernel_launch(void* const* d_in, const int* in_sizes, int n_in,
                              void* d_out, int out_size, void* d_ws, size_t ws_size,
                              hipStream_t stream) {
    const float* x   = (const float*)d_in[0];
    const float* Wih = (const float*)d_in[1];
    const float* Whh = (const float*)d_in[2];
    const float* bih = (const float*)d_in[3];
    const float* bhh = (const float*)d_in[4];
    const float* h0  = (const float*)d_in[5];
    const float* c0  = (const float*)d_in[6];

    float* out = (float*)d_out;                          // [B,T,H]
    float* hN  = out + (size_t)Bsz * Tlen * Hdim;        // [B,H]
    float* cN  = hN + (size_t)Bsz * Hdim;                // [B,H]

    // workspace carve (aligned offsets)
    const size_t xg_bytes  = (size_t)M1 * G4 * sizeof(float);     // 536,870,912
    const size_t whh_bytes = (size_t)G4 * Hdim * sizeof(__bf16);  //   8,388,608
    const size_t hg_bytes  = (size_t)2 * Bsz * Hdim * sizeof(__bf16); // 65,536
    char* base = (char*)d_ws;
    float*    xg_ws  = (float*)base;
    __bf16*   Whh_bf = (__bf16*)(base + xg_bytes);
    __bf16*   hg     = (__bf16*)(base + xg_bytes + whh_bytes);
    unsigned* cnt    = (unsigned*)(base + xg_bytes + whh_bytes + hg_bytes);

    // Phase 0
    lstm_init_kernel<<<(G4 * Hdim + 255) / 256, 256, 0, stream>>>(Whh, h0, Whh_bf, hg, cnt);

    // Phase 1: xg GEMM  (grid: N/64 x M/64)
    dim3 g1(G4 / TN, M1 / TM);
    gemm_xg_kernel<<<g1, 256, 0, stream>>>(x, Wih, bih, bhh, xg_ws);

    // Phase 2: persistent scan (16 workgroups == one cluster's worth)
    lstm_scan_kernel<<<Bsz, 256, 0, stream>>>(xg_ws, Whh_bf, c0, out, hN, cN, hg, cnt);
}
